// MoeLayer_27590869909647
// MI455X (gfx1250) — matrix operand order repaired and checked by reading, compiled, not run on hardware
//
#include <hip/hip_runtime.h>
#include <hip/hip_bf16.h>
#include <math.h>

typedef __attribute__((ext_vector_type(16))) _Float16 v16h;
typedef __attribute__((ext_vector_type(8)))  float    v8f;

#define B_   4
#define S_   2048
#define D_   1024
#define F_   4096
#define E_   8
#define K_   2
#define T_   64
#define NTOK (B_ * S_)   /* 8192 tokens */
#define TM   32          /* tokens per FFN tile = 2 WMMA M-tiles */
#define FC   128         /* F-chunk held in LDS between the two GEMMs */

/* packed-weight geometry: [E][N/16 panels][K/32 kblocks][32 lanes][16 f16] */
#define P1   (F_ / 16)   /* W1 panels = 256 */
#define KB1  (D_ / 32)   /* W1 kblocks = 32 */
#define P2   (D_ / 16)   /* W2 panels = 64  */
#define KB2  (F_ / 32)   /* W2 kblocks = 128 */

/* ---------------- helpers ---------------- */

__device__ __forceinline__ float gelu_tanh(float v) {
    const float c = 0.7978845608028654f;
    float t = tanhf(c * (v + 0.044715f * v * v * v));
    return 0.5f * v * (1.0f + t);
}

/* A-fragment (16x32 f16, M x K) from an LDS tile with row stride `stride`,
   rows [mbase, mbase+16).
   lanes 0-15 -> rows M=0..15 with K = 0..15;
               lanes 16-31 -> same rows with K = 16..31 (matches packed B). */
__device__ __forceinline__ v16h load_a_frag(const _Float16* Sbase, int stride,
                                            int mbase, int kbase, int lane) {
    int m     = mbase + (lane & 15);
    int khalf = (lane >> 4) ? 16 : 0;
    const _Float16* p = Sbase + m * stride + kbase + khalf;
    v16h a;
#pragma unroll
    for (int i = 0; i < 8; ++i) a[i] = p[i];
#pragma unroll
    for (int i = 0; i < 8; ++i) a[8 + i] = p[8 + i];
    return a;
}

/* B-fragment from pre-swizzled f16 weights: one contiguous 32B load per lane. */
__device__ __forceinline__ v16h load_b_frag_p(const _Float16* Wpe, int KB,
                                              int panel, int kb, int lane) {
    size_t off = (((size_t)panel * KB + kb) * 32 + lane) * 16;
    return *(const v16h*)(Wpe + off);
}

/* ---------------- kernel 0: zero output + aux counters ---------------- */

__global__ void moe_zero(float* __restrict__ out, int n_out,
                         int* __restrict__ cnt, float* __restrict__ sum_sm) {
    int i = blockIdx.x * blockDim.x + threadIdx.x;
    if (i < n_out) out[i] = 0.0f;
    if (i < E_) { cnt[i] = 0; sum_sm[i] = 0.0f; }
}

/* ---------------- kernel 0b: pack f32 weights -> f16 fragment order ------
   run r = ((e*P + p)*KB + kb)*32 + lane ; each run emits 16 contiguous f16:
   element j of lane l is W[e][K = kb*32 + (l>=16?16:0) + j][N = p*16 + l%16] */

__global__ void moe_pack_w(const float* __restrict__ W, _Float16* __restrict__ Wp,
                           int Kdim, int Ndim) {
    const int KB = Kdim / 32, P = Ndim / 16;
    size_t runs = (size_t)E_ * P * KB * 32;
    size_t r = (size_t)blockIdx.x * blockDim.x + threadIdx.x;
    if (r >= runs) return;
    int l = (int)(r & 31);
    size_t t = r >> 5;
    int kb = (int)(t % KB); t /= KB;
    int p  = (int)(t % P);  t /= P;
    int e  = (int)t;
    int n  = p * 16 + (l & 15);
    int k0 = kb * 32 + ((l >> 4) ? 16 : 0);
    const float* src = W + ((size_t)e * Kdim + k0) * Ndim + n;
    _Float16* dst = Wp + r * 16;
#pragma unroll
    for (int j = 0; j < 16; ++j) dst[j] = (_Float16)src[(size_t)j * Ndim];
}

/* ---------------- kernel 1: gate + top-2 routing + aux stats ---------------- */

__global__ void moe_gate_route(const float* __restrict__ x,
                               const float* __restrict__ task,
                               const float* __restrict__ alpha_p,
                               const float* __restrict__ Wg_in,
                               const float* __restrict__ bg_in,
                               const float* __restrict__ Wg_task,
                               const float* __restrict__ bg_task,
                               int* __restrict__ cnt,
                               float* __restrict__ sum_sm,
                               int* __restrict__ tok_idx,
                               float* __restrict__ tok_w) {
    int tok = blockIdx.x * blockDim.x + threadIdx.x;
    if (tok >= NTOK) return;
    float alpha = alpha_p[0];

    float accx[E_], acct[E_];
#pragma unroll
    for (int e = 0; e < E_; ++e) { accx[e] = 0.0f; acct[e] = 0.0f; }

    const float* xr = x + (size_t)tok * D_;
    for (int d = 0; d < D_; ++d) {
        float xv = xr[d];
        const float* wr = Wg_in + (size_t)d * E_;
#pragma unroll
        for (int e = 0; e < E_; ++e) accx[e] = fmaf(xv, wr[e], accx[e]);
    }
    const float* tr = task + (size_t)tok * T_;
    for (int t = 0; t < T_; ++t) {
        float tv = tr[t];
        const float* wr = Wg_task + (size_t)t * E_;
#pragma unroll
        for (int e = 0; e < E_; ++e) acct[e] = fmaf(tv, wr[e], acct[e]);
    }

    float lg[E_], mx = -1e30f;
#pragma unroll
    for (int e = 0; e < E_; ++e) {
        lg[e] = (1.0f - alpha) * (accx[e] + bg_in[e]) + alpha * (acct[e] + bg_task[e]);
        mx = fmaxf(mx, lg[e]);
    }
    float Z = 0.0f;
#pragma unroll
    for (int e = 0; e < E_; ++e) Z += __expf(lg[e] - mx);
#pragma unroll
    for (int e = 0; e < E_; ++e) atomicAdd(&sum_sm[e], __expf(lg[e] - mx) / Z);

    /* top-2 (descending, first occurrence on ties, matching top_k) */
    int i0 = 0;
#pragma unroll
    for (int e = 1; e < E_; ++e) if (lg[e] > lg[i0]) i0 = e;
    int i1 = -1;
#pragma unroll
    for (int e = 0; e < E_; ++e) {
        if (e == i0) continue;
        if (i1 < 0 || lg[e] > lg[i1]) i1 = e;
    }
    float w0 = 1.0f / (1.0f + __expf(lg[i1] - lg[i0]));
    float w1 = 1.0f - w0;

    int p0 = atomicAdd(&cnt[i0], 1);
    tok_idx[i0 * NTOK + p0] = tok;  tok_w[i0 * NTOK + p0] = w0;
    int p1 = atomicAdd(&cnt[i1], 1);
    tok_idx[i1 * NTOK + p1] = tok;  tok_w[i1 * NTOK + p1] = w1;
}

/* ---------------- kernel 2: fused expert FFN (WMMA) ----------------
   Block = 32 routed tokens of one expert, 256 threads = 8 wave32s.
   Each wave owns 128 output columns for both 16-row halves:
   2 x 8 v8f accumulators live in VGPRs for the whole F loop.
   B-fragments: single 32B/lane loads from pre-swizzled f16 weights.
   Dynamic LDS: Xs 64KB (32x1024 f16) + Hs 8KB (32x128 f16) + meta.  */

__global__ void __launch_bounds__(256)
moe_ffn(const float* __restrict__ x,
        const _Float16* __restrict__ W1p, const float* __restrict__ b1,
        const _Float16* __restrict__ W2p, const float* __restrict__ b2,
        const int*   __restrict__ cnt,
        const int*   __restrict__ tok_idx,
        const float* __restrict__ tok_w,
        float* __restrict__ out) {
    const int ex   = blockIdx.y;       /* expert */
    const int tile = blockIdx.x;
    const int n    = cnt[ex];
    const int row0 = tile * TM;
    if (row0 >= n) return;

    extern __shared__ __align__(16) char smem[];
    _Float16* Xs    = (_Float16*)smem;                         /* TM x D_  */
    _Float16* Hs    = (_Float16*)(smem + TM * D_ * 2);         /* TM x FC  */
    int*      s_tok = (int*)  (smem + TM * D_ * 2 + TM * FC * 2);
    float*    s_w   = (float*)(smem + TM * D_ * 2 + TM * FC * 2 + TM * 4);

    const int wave = threadIdx.x >> 5;
    const int lane = threadIdx.x & 31;

    if (threadIdx.x < TM) {
        int r = row0 + threadIdx.x;
        if (r < n) {
            s_tok[threadIdx.x] = tok_idx[ex * NTOK + r];
            s_w[threadIdx.x]   = tok_w[ex * NTOK + r];
        } else {                       /* pad rows: valid address, zero weight */
            s_tok[threadIdx.x] = tok_idx[ex * NTOK + row0];
            s_w[threadIdx.x]   = 0.0f;
        }
    }
    __syncthreads();

    for (int idx = threadIdx.x; idx < TM * D_; idx += 256) {
        int r = idx >> 10;             /* / D_ */
        int c = idx & (D_ - 1);
        Xs[idx] = (_Float16)x[(size_t)s_tok[r] * D_ + c];
    }
    __syncthreads();

    const _Float16* W1pe = W1p + (size_t)ex * P1 * KB1 * 512;  /* 512 f16 per (panel,kb) */
    const _Float16* W2pe = W2p + (size_t)ex * P2 * KB2 * 512;

    v8f accY0[8] = {};                 /* rows 0-15, wave's 128 columns  */
    v8f accY1[8] = {};                 /* rows 16-31                     */

    for (int f0 = 0; f0 < F_; f0 += FC) {
        /* phase 1: this wave's H tile [32 x 16]: X(32x1024) @ W1[:, hcol..hcol+16) */
        v8f hacc0 = {};
        v8f hacc1 = {};
        const int hcol   = f0 + wave * 16;
        const int hpanel = (f0 >> 4) + wave;
        for (int kb = 0; kb < KB1; ++kb) {
            v16h b  = load_b_frag_p(W1pe, KB1, hpanel, kb, lane);
            v16h a0 = load_a_frag(Xs, D_, 0,  kb * 32, lane);
            v16h a1 = load_a_frag(Xs, D_, 16, kb * 32, lane);
            hacc0 = __builtin_amdgcn_wmma_f32_16x16x32_f16(
                false, a0, false, b, (short)0, hacc0, false, false);
            hacc1 = __builtin_amdgcn_wmma_f32_16x16x32_f16(
                false, a1, false, b, (short)0, hacc1, false, false);
        }
        {
            const int m0   = (lane >> 4) ? 8 : 0;
            const int ncol = lane & 15;
            const float bias1 = b1[ex * F_ + hcol + ncol];
#pragma unroll
            for (int r = 0; r < 8; ++r) {
                float g0 = gelu_tanh(hacc0[r] + bias1);
                float g1 = gelu_tanh(hacc1[r] + bias1);
                Hs[(m0 + r) * FC + wave * 16 + ncol]      = (_Float16)g0;
                Hs[(16 + m0 + r) * FC + wave * 16 + ncol] = (_Float16)g1;
            }
        }
        __syncthreads();

        /* phase 2: Y[:, wave*128 .. +128) += H(32x128) @ W2[f0..f0+128, cols) */
#pragma unroll
        for (int ks = 0; ks < FC; ks += 32) {
            const int kb2 = (f0 + ks) >> 5;
            v16h a0 = load_a_frag(Hs, FC, 0,  ks, lane);
            v16h a1 = load_a_frag(Hs, FC, 16, ks, lane);
#pragma unroll
            for (int nt = 0; nt < 8; ++nt) {
                v16h bf = load_b_frag_p(W2pe, KB2, wave * 8 + nt, kb2, lane);
                accY0[nt] = __builtin_amdgcn_wmma_f32_16x16x32_f16(
                    false, a0, false, bf, (short)0, accY0[nt], false, false);
                accY1[nt] = __builtin_amdgcn_wmma_f32_16x16x32_f16(
                    false, a1, false, bf, (short)0, accY1[nt], false, false);
            }
        }
        __syncthreads();
    }

    /* epilogue: +b2, scale by combine weight, scatter-add (token hit by K=2 experts) */
    {
        const int m0   = (lane >> 4) ? 8 : 0;
        const int ncol = lane & 15;
#pragma unroll
        for (int nt = 0; nt < 8; ++nt) {
            const int col  = wave * 128 + nt * 16 + ncol;
            const float b2v = b2[ex * D_ + col];
#pragma unroll
            for (int r = 0; r < 8; ++r) {
                const int mA = m0 + r;
                const int mB = 16 + m0 + r;
                atomicAdd(&out[(size_t)s_tok[mA] * D_ + col],
                          s_w[mA] * (accY0[nt][r] + b2v));
                atomicAdd(&out[(size_t)s_tok[mB] * D_ + col],
                          s_w[mB] * (accY1[nt][r] + b2v));
            }
        }
    }
}

/* ---------------- kernel 3: l_aux ---------------- */

__global__ void moe_finalize(const int* __restrict__ cnt,
                             const float* __restrict__ sum_sm,
                             float* __restrict__ out_laux) {
    if (threadIdx.x == 0) {
        float acc = 0.0f;
        for (int e = 0; e < E_; ++e)
            acc += (sum_sm[e] / (float)NTOK) * ((float)cnt[e] / (float)NTOK);
        /* mean(avg_w * avg_c) * E == sum_e(avg_w * avg_c) */
        out_laux[0] = acc;
    }
}

/* ---------------- launch ---------------- */

extern "C" void kernel_launch(void* const* d_in, const int* in_sizes, int n_in,
                              void* d_out, int out_size, void* d_ws, size_t ws_size,
                              hipStream_t stream) {
    const float* x       = (const float*)d_in[0];
    const float* task    = (const float*)d_in[1];
    const float* alpha   = (const float*)d_in[2];
    const float* Wg_in   = (const float*)d_in[3];
    const float* bg_in   = (const float*)d_in[4];
    const float* Wg_task = (const float*)d_in[5];
    const float* bg_task = (const float*)d_in[6];
    const float* W1      = (const float*)d_in[7];
    const float* b1      = (const float*)d_in[8];
    const float* W2      = (const float*)d_in[9];
    const float* b2      = (const float*)d_in[10];
    float* out = (float*)d_out;

    /* workspace layout (~129 MB):
       [0,32)            cnt[E] (int)
       [256,288)         sum_sm[E] (float)
       [512, +256KB)     tok_idx[E*NTOK] (int)
       [.., +256KB)      tok_w[E*NTOK] (float)
       [1MB, +64MB)      W1 packed f16 (fragment order)
       [65MB, +64MB)     W2 packed f16 (fragment order)                */
    char*      ws      = (char*)d_ws;
    int*       cnt     = (int*)ws;
    float*     sum_sm  = (float*)(ws + 256);
    int*       tok_idx = (int*)(ws + 512);
    float*     tok_w   = (float*)(ws + 512 + (size_t)E_ * NTOK * sizeof(int));
    _Float16*  W1p     = (_Float16*)(ws + (1u << 20));
    _Float16*  W2p     = (_Float16*)(ws + (1u << 20) + (size_t)E_ * D_ * F_ * 2);
    (void)in_sizes; (void)n_in; (void)out_size; (void)ws_size;

    const int n_out = NTOK * D_;
    moe_zero<<<(n_out + 255) / 256, 256, 0, stream>>>(out, n_out, cnt, sum_sm);

    /* pack weights to f16 fragment order (runs once per launch, HBM-trivial) */
    {
        size_t runs = (size_t)E_ * P1 * KB1 * 32;   /* == E*D*F/16 for both */
        moe_pack_w<<<(unsigned)((runs + 255) / 256), 256, 0, stream>>>(W1, W1p, D_, F_);
        moe_pack_w<<<(unsigned)((runs + 255) / 256), 256, 0, stream>>>(W2, W2p, F_, D_);
    }

    moe_gate_route<<<(NTOK + 255) / 256, 256, 0, stream>>>(
        x, task, alpha, Wg_in, bg_in, Wg_task, bg_task,
        cnt, sum_sm, tok_idx, tok_w);

    /* x = tile index (fast-varying -> same-expert tiles launch together, L2 reuse),
       y = expert. NTOK/TM tiles covers the worst case cnt[e] == NTOK.
       Dynamic LDS: 32x1024 f16 X tile + 32x128 f16 H tile + 32 (int+float) meta. */
    const size_t smem_bytes = (size_t)TM * D_ * 2 + (size_t)TM * FC * 2
                            + (size_t)TM * (sizeof(int) + sizeof(float));
    dim3 grid(NTOK / TM, E_);
    moe_ffn<<<grid, 256, smem_bytes, stream>>>(x, W1p, b1, W2p, b2,
                                               cnt, tok_idx, tok_w, out);

    moe_finalize<<<1, 32, 0, stream>>>(cnt, sum_sm, out + n_out);
}
